// HPOModel_37821482009110
// MI455X (gfx1250) — matrix operand order repaired and checked by reading, not compiled, not run on hardware
//
#include <hip/hip_runtime.h>
#include <stdint.h>

typedef __attribute__((ext_vector_type(16))) __bf16 bf16x16;
typedef __attribute__((ext_vector_type(8)))  __bf16 bf16x8;
typedef __attribute__((ext_vector_type(8)))  float  f32x8;
typedef __attribute__((ext_vector_type(4)))  int    v4i;

#define B_  2048
#define S_  32
#define E_  1024
#define O_  1024
#define D_  1024
#define NC_ 15001     // N_CONCEPT + 1
#define EPS_ 1e-12f

#if defined(__has_builtin)
#if __has_builtin(__builtin_amdgcn_global_load_async_to_lds_b128)
#define HAVE_ASYNC_LDS 1
#endif
#endif

union FragU { bf16x16 v; bf16x8 h[2]; };

// ---- fragment loaders (work for both global bf16 and LDS bf16 pointers) ---
// A-fragment: lane holds row (L&15); K = {0..7} at p, {16..23} at p+16
// (p already includes kbA = (lane<16)?0:8)
__device__ __forceinline__ bf16x16 loadA_bf(const __bf16* p) {
  FragU f;
  f.h[0] = *(const bf16x8*)(p);
  f.h[1] = *(const bf16x8*)(p + 16);
  return f.v;
}
// B-fragment: lane holds column (L&15); K = {0..7} at p, {16..23} at p+16
// (p already includes kbB = (lane<16)?0:8)
__device__ __forceinline__ bf16x16 loadB_bf(const __bf16* p) {
  FragU f;
  f.h[0] = *(const bf16x8*)(p);
  f.h[1] = *(const bf16x8*)(p + 16);
  return f.v;
}

// A-fragment converted on the fly from f32 (kernel 1 only) -> v_cvt_pk_bf16_f32
__device__ __forceinline__ bf16x16 cvtA_f32(const float* __restrict__ p) {
  float4 a = ((const float4*)p)[0];
  float4 b = ((const float4*)p)[1];
  float4 c = ((const float4*)(p + 16))[0];
  float4 d = ((const float4*)(p + 16))[1];
  bf16x16 v;
  v[0]  = (__bf16)a.x; v[1]  = (__bf16)a.y; v[2]  = (__bf16)a.z; v[3]  = (__bf16)a.w;
  v[4]  = (__bf16)b.x; v[5]  = (__bf16)b.y; v[6]  = (__bf16)b.z; v[7]  = (__bf16)b.w;
  v[8]  = (__bf16)c.x; v[9]  = (__bf16)c.y; v[10] = (__bf16)c.z; v[11] = (__bf16)c.w;
  v[12] = (__bf16)d.x; v[13] = (__bf16)d.y; v[14] = (__bf16)d.z; v[15] = (__bf16)d.w;
  return v;
}

__device__ __forceinline__ f32x8 wmma_bf16(bf16x16 a, bf16x16 b, f32x8 c) {
  return __builtin_amdgcn_wmma_f32_16x16x32_bf16(false, a, false, b,
                                                 (short)0, c, false, false);
}

// ---- async global->LDS copy (16B), with graceful fallback -----------------
// Builtin signature (from hipcc diagnostic): param0 = v4i in AS1 (global src),
// param1 = LDS dst, then imm offset / cpol.
__device__ __forceinline__ void copy16_async(const __bf16* g, __bf16* l) {
#ifdef HAVE_ASYNC_LDS
  typedef __attribute__((address_space(1))) v4i gv4i;
  typedef __attribute__((address_space(3))) v4i lv4i;
  __builtin_amdgcn_global_load_async_to_lds_b128((gv4i*)(v4i*)g,
                                                 (lv4i*)(v4i*)l, 0, 0);
#else
  *(bf16x8*)l = *(const bf16x8*)g;
#endif
}
__device__ __forceinline__ void async_wait_all() {
#ifdef HAVE_ASYNC_LDS
#if __has_builtin(__builtin_amdgcn_s_wait_asynccnt)
  __builtin_amdgcn_s_wait_asynccnt(0);
#else
  asm volatile("s_wait_asynccnt 0x0" ::: "memory");
#endif
#endif
}

// ================= f32 -> bf16 bulk conversion (memory-bound pass) =========
__global__ __launch_bounds__(256) void k_cvt_bf16(const float* __restrict__ src,
                                                  __bf16* __restrict__ dst,
                                                  long n) {
  const size_t i = ((size_t)blockIdx.x * blockDim.x + threadIdx.x) * 8;
  if (i >= (size_t)n) return;
  float4 a = ((const float4*)(src + i))[0];
  float4 b = ((const float4*)(src + i))[1];
  bf16x8 r;
  r[0] = (__bf16)a.x; r[1] = (__bf16)a.y; r[2] = (__bf16)a.z; r[3] = (__bf16)a.w;
  r[4] = (__bf16)b.x; r[5] = (__bf16)b.y; r[6] = (__bf16)b.z; r[7] = (__bf16)b.w;
  *(bf16x8*)(dst + i) = r;
}

// ================= Kernel 1: conv1d(ks=1) + bias + relu + max over S =======
// grid (O/256, B), block 256; wave = 1 s-tile x 4 o-tiles (32x256 per block)
__global__ __launch_bounds__(256) void k_conv_max(
    const float* __restrict__ data, const __bf16* __restrict__ conv_wB,
    const float* __restrict__ conv_b, __bf16* __restrict__ Mout) {
  __shared__ int smax[256];
  const int tid  = threadIdx.x;
  const int lane = tid & 31;
  const int wave = tid >> 5;
  const int b     = blockIdx.y;
  const int mtile = wave & 1;
  const int ngrp  = wave >> 1;                  // 0..3, each covers 64 cols
  const int l15   = lane & 15;
  const int kbA   = (lane < 16) ? 0 : 8;
  const int kbB   = (lane < 16) ? 0 : 8;

  const float* aptr = data + ((size_t)b * S_ + (size_t)(mtile * 16 + l15)) * E_ + kbA;
  const __bf16* bp[4];
#pragma unroll
  for (int t = 0; t < 4; ++t)
    bp[t] = conv_wB +
            (size_t)(blockIdx.x * 256 + ngrp * 64 + t * 16 + l15) * E_ + kbB;

  f32x8 acc[4] = {{}, {}, {}, {}};
  for (int k0 = 0; k0 < E_; k0 += 32) {
    bf16x16 af = cvtA_f32(aptr + k0);
#pragma unroll
    for (int t = 0; t < 4; ++t)
      acc[t] = wmma_bf16(af, loadB_bf(bp[t] + k0), acc[t]);
  }

  smax[tid] = 0;
  __syncthreads();

#pragma unroll
  for (int t = 0; t < 4; ++t) {
    const int nloc = ngrp * 64 + t * 16 + l15;
    const float bias = conv_b[blockIdx.x * 256 + nloc];
    float m = 0.0f;
#pragma unroll
    for (int v = 0; v < 8; ++v)
      m = fmaxf(m, fmaxf(acc[t][v] + bias, 0.0f));  // relu>=0, init 0 safe
    m = fmaxf(m, __shfl_xor(m, 16));  // rows 0-7 vs 8-15 of the tile
    if (lane < 16) atomicMax(&smax[nloc], __float_as_int(m));
  }
  __syncthreads();
  Mout[(size_t)b * O_ + blockIdx.x * 256 + tid] = (__bf16)__int_as_float(smax[tid]);
}

// ================= Kernel 2: linear + bias + relu, row sum-sq ==============
// grid (D/256, B/32), block 256
__global__ __launch_bounds__(256) void k_linear_norm(
    const __bf16* __restrict__ MinB, const __bf16* __restrict__ lin_wB,
    const float* __restrict__ lin_b, __bf16* __restrict__ X,
    float* __restrict__ sumsq) {
  const int tid  = threadIdx.x;
  const int lane = tid & 31;
  const int wave = tid >> 5;
  const int rbase = blockIdx.y * 32 + (wave & 1) * 16;
  const int nbase = blockIdx.x * 256 + (wave >> 1) * 64;
  const int l15  = lane & 15;
  const int kbA  = (lane < 16) ? 0 : 8;
  const int kbB  = (lane < 16) ? 0 : 8;

  const __bf16* aptr = MinB + (size_t)(rbase + l15) * O_ + kbA;
  const __bf16* bp[4];
#pragma unroll
  for (int t = 0; t < 4; ++t)
    bp[t] = lin_wB + (size_t)(nbase + t * 16 + l15) * O_ + kbB;

  f32x8 acc[4] = {{}, {}, {}, {}};
  for (int k0 = 0; k0 < O_; k0 += 32) {
    bf16x16 af = loadA_bf(aptr + k0);
#pragma unroll
    for (int t = 0; t < 4; ++t)
      acc[t] = wmma_bf16(af, loadB_bf(bp[t] + k0), acc[t]);
  }

  float bias[4];
#pragma unroll
  for (int t = 0; t < 4; ++t) bias[t] = lin_b[nbase + t * 16 + l15];

  const int rhalf = (lane >= 16) ? 8 : 0;
#pragma unroll
  for (int v = 0; v < 8; ++v) {
    const int row = rbase + v + rhalf;
    float s = 0.0f;
#pragma unroll
    for (int t = 0; t < 4; ++t) {
      float x = fmaxf(acc[t][v] + bias[t], 0.0f);
      X[(size_t)row * D_ + nbase + t * 16 + l15] = (__bf16)x;
      s += x * x;
    }
    // all 4 tiles share `row`: one reduction over the 16-lane half
    s += __shfl_xor(s, 1); s += __shfl_xor(s, 2);
    s += __shfl_xor(s, 4); s += __shfl_xor(s, 8);
    if (l15 == 0) atomicAdd(&sumsq[row], s);
  }
}

// ================= small init / finalize kernels ===========================
__global__ void k_init_norm(float* __restrict__ sumsq) {
  int i = blockIdx.x * blockDim.x + threadIdx.x;
  if (i < B_) sumsq[i] = 0.0f;
}

__global__ void k_rscale(float* __restrict__ sumsq) {
  int i = blockIdx.x * blockDim.x + threadIdx.x;
  if (i < B_) {
    float n = sqrtf(sumsq[i]);
    sumsq[i] = 1.0f / fmaxf(n, EPS_);   // normalize folded into final GEMM
  }
}

__global__ void k_init_graph(const float* __restrict__ gcn_bias,
                             float* __restrict__ graph) {
  size_t i = (size_t)blockIdx.x * blockDim.x + threadIdx.x;
  if (i < (size_t)NC_ * D_) graph[i] = gcn_bias[i & (D_ - 1)];
}

// ================= Kernel 3: sparse scatter  graph[r] += v * H0[c] =========
// one block per nnz; H0 (61 MB) and graph (61 MB) are L2-resident (192 MB L2)
__global__ __launch_bounds__(256) void k_gcn_scatter(
    const int* __restrict__ rows, const int* __restrict__ cols,
    const float* __restrict__ vals, const float* __restrict__ H0,
    float* __restrict__ graph, int nnz) {
  const int i = blockIdx.x;
  if (i >= nnz) return;
  const int r = rows[i];
  const int c = cols[i];
  const float v = vals[i];
  const float4 h = ((const float4*)(H0 + (size_t)c * D_))[threadIdx.x];
  float* dst = graph + (size_t)r * D_ + threadIdx.x * 4;
  atomicAdd(dst + 0, v * h.x);
  atomicAdd(dst + 1, v * h.y);
  atomicAdd(dst + 2, v * h.z);
  atomicAdd(dst + 3, v * h.w);
}

// ================= Kernel 4: logits = diag(rscale) * X @ graph^T ===========
// grid (ceil(NC/256), B/32), block 256.
// The 32x1024 bf16 A-panel (64 KB) is staged once per block into LDS with
// async global->LDS copies; the K-loop reads A from LDS, streams B from L2.
__global__ __launch_bounds__(256) void k_logits(
    const __bf16* __restrict__ X, const __bf16* __restrict__ graphB,
    const float* __restrict__ rscale, float* __restrict__ out) {
  __shared__ __align__(16) __bf16 Alds[32 * D_];   // 64 KB of 320 KB WGP LDS
  const int tid  = threadIdx.x;
  const int lane = tid & 31;
  const int wave = tid >> 5;
  const int rbase0 = blockIdx.y * 32;
  const int rbase  = rbase0 + (wave & 1) * 16;
  const int nbase  = blockIdx.x * 256 + (wave >> 1) * 64;
  const int l15  = lane & 15;
  const int kbA  = (lane < 16) ? 0 : 8;
  const int kbB  = (lane < 16) ? 0 : 8;

  // ---- stage A panel: 32 rows x 1024 K of bf16 = 4096 x 16B chunks ----
  {
    const __bf16* gA = X + (size_t)rbase0 * D_;
#pragma unroll
    for (int c = 0; c < 16; ++c)
      copy16_async(gA + (size_t)(c * 256 + tid) * 8, Alds + (size_t)(c * 256 + tid) * 8);
    async_wait_all();
    __syncthreads();
  }

  int ncol[4], ncl[4];
  const __bf16* bp[4];
#pragma unroll
  for (int t = 0; t < 4; ++t) {
    ncol[t] = nbase + t * 16 + l15;
    ncl[t]  = (ncol[t] < NC_) ? ncol[t] : (NC_ - 1);   // clamp B rows for loads
    bp[t]   = graphB + (size_t)ncl[t] * D_ + kbB;
  }
  const __bf16* al = Alds + (size_t)((wave & 1) * 16 + l15) * D_ + kbA;

  f32x8 acc[4] = {{}, {}, {}, {}};
  for (int k0 = 0; k0 < D_; k0 += 32) {
    bf16x16 af = loadA_bf(al + k0);      // ds_load_b128 x2
#pragma unroll
    for (int t = 0; t < 4; ++t)
      acc[t] = wmma_bf16(af, loadB_bf(bp[t] + k0), acc[t]);
  }

  const int rhalf = (lane >= 16) ? 8 : 0;
#pragma unroll
  for (int v = 0; v < 8; ++v) {
    const int row = rbase + v + rhalf;
    const float rs = rscale[row];
#pragma unroll
    for (int t = 0; t < 4; ++t)
      if (ncol[t] < NC_) out[(size_t)row * NC_ + ncol[t]] = acc[t][v] * rs;
  }
}

// ===========================================================================
extern "C" void kernel_launch(void* const* d_in, const int* in_sizes, int n_in,
                              void* d_out, int out_size, void* d_ws, size_t ws_size,
                              hipStream_t stream) {
  const float* data   = (const float*)d_in[0];
  // d_in[1] = seq_len : unused by the reference computation
  const float* conv_w = (const float*)d_in[2];
  const float* conv_b = (const float*)d_in[3];
  const float* lin_w  = (const float*)d_in[4];
  const float* lin_b  = (const float*)d_in[5];
  const float* gcn_b  = (const float*)d_in[6];
  const float* H0     = (const float*)d_in[7];
  const int*   idx    = (const int*)d_in[8];
  const float* vals   = (const float*)d_in[9];
  const int nnz = in_sizes[9];

  // ---- workspace layout (all sizes 256B-aligned) ----
  char* ws = (char*)d_ws;
  size_t off = 0;
  float*  graphF = (float*)(ws + off);  off += (size_t)NC_ * D_ * 4;   // 61.4 MB
  __bf16* graphB = (__bf16*)(ws + off); off += (size_t)NC_ * D_ * 2;   // 30.7 MB
  __bf16* convwB = (__bf16*)(ws + off); off += (size_t)O_ * E_ * 2;    //  2  MB
  __bf16* linwB  = (__bf16*)(ws + off); off += (size_t)D_ * O_ * 2;    //  2  MB
  __bf16* MoutB  = (__bf16*)(ws + off); off += (size_t)B_ * O_ * 2;    //  4  MB
  __bf16* XB     = (__bf16*)(ws + off); off += (size_t)B_ * D_ * 2;    //  4  MB
  float*  nbuf   = (float*)(ws + off);  off += (size_t)B_ * 4;

  // ---- one-time weight conversions (memory-bound, tiny) ----
  {
    const long nw = (long)O_ * E_;
    k_cvt_bf16<<<(unsigned)((nw / 8 + 255) / 256), 256, 0, stream>>>(conv_w, convwB, nw);
    k_cvt_bf16<<<(unsigned)((nw / 8 + 255) / 256), 256, 0, stream>>>(lin_w, linwB, nw);
  }
  k_init_norm<<<(B_ + 255) / 256, 256, 0, stream>>>(nbuf);
  {
    const size_t total = (size_t)NC_ * D_;
    k_init_graph<<<(unsigned)((total + 255) / 256), 256, 0, stream>>>(gcn_b, graphF);
  }

  // 1) conv(ks=1) + relu + max over S -> MoutB (bf16)
  k_conv_max<<<dim3(O_ / 256, B_), 256, 0, stream>>>(data, convwB, conv_b, MoutB);

  // 2) linear + relu -> XB (bf16), accumulate row sum-sq -> nbuf
  k_linear_norm<<<dim3(D_ / 256, B_ / 32), 256, 0, stream>>>(MoutB, linwB, lin_b,
                                                             XB, nbuf);
  k_rscale<<<(B_ + 255) / 256, 256, 0, stream>>>(nbuf);

  // 3) sparse Laplacian GCN in f32, then one bulk cvt to bf16
  k_gcn_scatter<<<nnz, 256, 0, stream>>>(idx, idx + nnz, vals, H0, graphF, nnz);
  {
    const long ng = (long)NC_ * D_;
    k_cvt_bf16<<<(unsigned)((ng / 8 + 255) / 256), 256, 0, stream>>>(graphF, graphB, ng);
  }

  // 4) logits = (X / ||X||) @ graph^T  (normalization folded into epilogue)
  k_logits<<<dim3((NC_ + 255) / 256, B_ / 32), 256, 0, stream>>>(XB, graphB, nbuf,
                                                                 (float*)d_out);
}
